// simpleGRU_11089605558766
// MI455X (gfx1250) — matrix operand order repaired and checked by reading, not compiled, not run on hardware
//
#include <hip/hip_runtime.h>
#include <math.h>

// Problem constants (match reference)
#define BB  256
#define TT  512
#define FF  64
#define HH  1024
#define TLL 64

typedef __attribute__((ext_vector_type(16))) _Float16 v16h;
typedef __attribute__((ext_vector_type(8)))  _Float16 v8h;
typedef __attribute__((ext_vector_type(8)))  float    v8f;

union AFrag { v16h v; v8h h2[2]; };

__device__ __forceinline__ v8f wmma_f16(v16h a, v16h b, v8f c) {
  // 8 args: (neg_a, A, neg_b, B, c_mod, C, reuse_a, reuse_b)
  return __builtin_amdgcn_wmma_f32_16x16x32_f16(false, a, false, b, (short)0, c,
                                                false, false);
}

__device__ __forceinline__ float sigmoidf_(float x) {
  return 1.0f / (1.0f + __expf(-x));
}

// ---------------------------------------------------------------------------
// One GRU time step: computes h_out = GRUCell(h_in, input)
//   ENC = true : input is x_t (f16), uses W_ih via WMMA over K=F=64
//   ENC = false: input is scalar outprev[b], rank-1 term with dec_W_ih
//
// Mapping (weight-sharing oriented): blockIdx.x owns ONE hidden-column tile
// (n0), so all 8 waves of the block read IDENTICAL W_hh rows each K-iter —
// the WGP$ dedups them into a single L2 fill (8x less weight traffic than a
// wave-per-n-tile mapping). Each wave owns a different 16-row batch tile and
// accumulates the three gate tiles (r,z,n) for its 16x16 h'-tile.
// ---------------------------------------------------------------------------
#define WAVES 8

template <bool ENC>
__global__ __launch_bounds__(WAVES * 32)
void gru_step(const _Float16* __restrict__ hin,
              _Float16*       __restrict__ hout,
              const _Float16* __restrict__ Whh,   // [3H,H] f16 row-major
              const _Float16* __restrict__ Wih,   // [3H,F] f16 row-major (ENC)
              const _Float16* __restrict__ xt,    // x_f16 + t*F       (ENC)
              const float*    __restrict__ bih,   // [3H]
              const float*    __restrict__ bhh,   // [3H]
              const float*    __restrict__ wdec,  // [3H]  dec W_ih    (!ENC)
              const float*    __restrict__ outprev) // [B]             (!ENC)
{
  const int lane = threadIdx.x & 31;
  const int wave = threadIdx.x >> 5;
  const int l    = lane & 15;
  const int hi   = lane >> 4;

  const int n0 = blockIdx.x * 16;                   // hidden-column tile (shared by block)
  const int m0 = (blockIdx.y * WAVES + wave) * 16;  // batch-row tile (per wave)

  v8f acc_r{}, acc_z{}, acc_hn{}, acc_in{};

  // ---- hidden-state GEMM: gates += h @ W_hh.T,  K = H ----
  // A fragment (16x32 f16): lane<16 row m0+l holds K k0..k0+15 (contiguous)
  //                         lane>=16 row m0+l holds K k0+16..k0+31
  const _Float16* ha = hin + (size_t)(m0 + l) * HH + hi * 16;
  // B fragment (32x16 f16): lane<16 col n0+l holds K k0..k0+15 (contiguous)
  //                         lane>=16 col n0+l holds K k0+16..k0+31
  const _Float16* w0 = Whh + ((size_t)(0 * HH + n0 + l)) * HH + hi * 16;
  const _Float16* w1 = Whh + ((size_t)(1 * HH + n0 + l)) * HH + hi * 16;
  const _Float16* w2 = Whh + ((size_t)(2 * HH + n0 + l)) * HH + hi * 16;

  // Warm the caches for this block's weight rows (L2-resident after step 0).
  __builtin_prefetch(w0, 0, 3);
  __builtin_prefetch(w1, 0, 3);
  __builtin_prefetch(w2, 0, 3);

  for (int k0 = 0; k0 < HH; k0 += 32) {
    AFrag a;
    a.h2[0] = *(const v8h*)(ha + k0);
    a.h2[1] = *(const v8h*)(ha + k0 + 8);
    v16h b0 = *(const v16h*)(w0 + k0);
    v16h b1 = *(const v16h*)(w1 + k0);
    v16h b2 = *(const v16h*)(w2 + k0);
    acc_r  = wmma_f16(a.v, b0, acc_r);
    acc_z  = wmma_f16(a.v, b1, acc_z);
    acc_hn = wmma_f16(a.v, b2, acc_hn);
  }

  // ---- encoder input GEMM: gates += x_t @ W_ih.T,  K = F = 64 ----
  if (ENC) {
    const _Float16* xa = xt + (size_t)(m0 + l) * (TT * FF) + hi * 16;
    const _Float16* u0 = Wih + ((size_t)(0 * HH + n0 + l)) * FF + hi * 16;
    const _Float16* u1 = Wih + ((size_t)(1 * HH + n0 + l)) * FF + hi * 16;
    const _Float16* u2 = Wih + ((size_t)(2 * HH + n0 + l)) * FF + hi * 16;
    for (int k0 = 0; k0 < FF; k0 += 32) {
      AFrag a;
      a.h2[0] = *(const v8h*)(xa + k0);
      a.h2[1] = *(const v8h*)(xa + k0 + 8);
      v16h b0 = *(const v16h*)(u0 + k0);
      v16h b1 = *(const v16h*)(u1 + k0);
      v16h b2 = *(const v16h*)(u2 + k0);
      acc_r  = wmma_f16(a.v, b0, acc_r);
      acc_z  = wmma_f16(a.v, b1, acc_z);
      acc_in = wmma_f16(a.v, b2, acc_in);
    }
  }

  // ---- GRU epilogue ----
  // C/D layout: lane L -> column n0+(L&15); VGPR i -> row m0 + i + 8*(L>=16)
  const int j = n0 + l;
  const float br_i = bih[j],       bz_i = bih[HH + j],     bn_i = bih[2 * HH + j];
  const float br_h = bhh[j],       bz_h = bhh[HH + j],     bn_h = bhh[2 * HH + j];
  float wdr = 0.f, wdz = 0.f, wdn = 0.f;
  if (!ENC) { wdr = wdec[j]; wdz = wdec[HH + j]; wdn = wdec[2 * HH + j]; }

#pragma unroll
  for (int i = 0; i < 8; ++i) {
    const int brow = m0 + hi * 8 + i;
    float op = 0.f;
    if (!ENC) op = outprev[brow];

    float rv  = acc_r[i] + br_i + br_h;
    float zv  = acc_z[i] + bz_i + bz_h;
    float inn = bn_i;
    if (ENC) inn += acc_in[i];
    else     { rv += op * wdr; zv += op * wdz; inn += op * wdn; }
    const float hn = acc_hn[i] + bn_h;

    const float r = sigmoidf_(rv);
    const float z = sigmoidf_(zv);
    const float n = tanhf(inn + r * hn);
    const float hold = (float)hin[(size_t)brow * HH + j];
    const float hnew = (1.0f - z) * n + z * hold;
    hout[(size_t)brow * HH + j] = (_Float16)hnew;
  }
}

// ---------------------------------------------------------------------------
// Decoder output projection: out[b] = h[b,:] . fc_W + fc_b
// One wave per batch row; 32-lane shuffle reduction.
// ---------------------------------------------------------------------------
__global__ __launch_bounds__(256)
void fc_out(const _Float16* __restrict__ h, const float* __restrict__ fcW,
            const float* __restrict__ fcb, float* __restrict__ outprev,
            float* __restrict__ dout, int t)
{
  const int lane = threadIdx.x & 31;
  const int wave = threadIdx.x >> 5;
  const int b = blockIdx.x * 8 + wave;

  float s = 0.f;
  for (int k = lane; k < HH; k += 32)
    s += (float)h[(size_t)b * HH + k] * fcW[k];
#pragma unroll
  for (int off = 16; off > 0; off >>= 1)
    s += __shfl_xor(s, off, 32);
  if (lane == 0) {
    const float o = s + fcb[0];
    outprev[b] = o;
    dout[(size_t)b * TLL + t] = o;
  }
}

// ---------------------------------------------------------------------------
// fp32 -> f16 conversion (weights / x), done once up front.
// ---------------------------------------------------------------------------
__global__ void cvt_f32_f16(const float* __restrict__ src,
                            _Float16* __restrict__ dst, int n)
{
  int i = blockIdx.x * blockDim.x + threadIdx.x;
  if (i < n) dst[i] = (_Float16)src[i];
}

// ---------------------------------------------------------------------------
extern "C" void kernel_launch(void* const* d_in, const int* in_sizes, int n_in,
                              void* d_out, int out_size, void* d_ws, size_t ws_size,
                              hipStream_t stream)
{
  (void)in_sizes; (void)n_in; (void)out_size; (void)ws_size;

  const float* x        = (const float*)d_in[0];
  const float* eWih     = (const float*)d_in[1];
  const float* eWhh     = (const float*)d_in[2];
  const float* eBih     = (const float*)d_in[3];
  const float* eBhh     = (const float*)d_in[4];
  const float* dWih     = (const float*)d_in[5];
  const float* dWhh     = (const float*)d_in[6];
  const float* dBih     = (const float*)d_in[7];
  const float* dBhh     = (const float*)d_in[8];
  const float* fcW      = (const float*)d_in[9];
  const float* fcb      = (const float*)d_in[10];
  float* out            = (float*)d_out;

  // Workspace layout (f16 copies + ping-pong hidden state + feedback scalar)
  char* ws = (char*)d_ws;
  const size_t nX    = (size_t)BB * TT * FF;          // 8,388,608
  const size_t nEWih = (size_t)3 * HH * FF;           // 196,608
  const size_t nWhh  = (size_t)3 * HH * HH;           // 3,145,728
  const size_t nH    = (size_t)BB * HH;               // 262,144

  _Float16* xf   = (_Float16*)ws;                 ws += nX    * 2;
  _Float16* eUf  = (_Float16*)ws;                 ws += nEWih * 2;
  _Float16* eWf  = (_Float16*)ws;                 ws += nWhh  * 2;
  _Float16* dWf  = (_Float16*)ws;                 ws += nWhh  * 2;
  _Float16* hA   = (_Float16*)ws;                 ws += nH    * 2;
  _Float16* hB   = (_Float16*)ws;                 ws += nH    * 2;
  float*    oprv = (float*)ws;                    ws += BB * sizeof(float);

  // ---- one-time prep: convert to f16, zero state ----
  {
    int thr = 256;
    cvt_f32_f16<<<(int)((nX    + thr - 1) / thr), thr, 0, stream>>>(x,    xf,  (int)nX);
    cvt_f32_f16<<<(int)((nEWih + thr - 1) / thr), thr, 0, stream>>>(eWih, eUf, (int)nEWih);
    cvt_f32_f16<<<(int)((nWhh  + thr - 1) / thr), thr, 0, stream>>>(eWhh, eWf, (int)nWhh);
    cvt_f32_f16<<<(int)((nWhh  + thr - 1) / thr), thr, 0, stream>>>(dWhh, dWf, (int)nWhh);
    hipMemsetAsync(hA,   0, nH * 2,             stream);
    hipMemsetAsync(oprv, 0, BB * sizeof(float), stream);
  }

  // grid.x = 64 n-tiles (block shares W rows), grid.y = 2 m-tile groups
  const dim3 grid(HH / 16, (BB / 16) / WAVES);
  const dim3 blk(WAVES * 32);

  // ---- encoder: 512 sequential GRU steps ----
  for (int t = 0; t < TT; ++t) {
    const _Float16* hi = (t & 1) ? hB : hA;
    _Float16*       ho = (t & 1) ? hA : hB;
    gru_step<true><<<grid, blk, 0, stream>>>(hi, ho, eWf, eUf, xf + (size_t)t * FF,
                                             eBih, eBhh, nullptr, nullptr);
  }
  // after 512 steps the state is back in hA

  // ---- decoder: 64 autoregressive steps ----
  for (int s = 0; s < TLL; ++s) {
    const _Float16* hi = (s & 1) ? hB : hA;
    _Float16*       ho = (s & 1) ? hA : hB;
    gru_step<false><<<grid, blk, 0, stream>>>(hi, ho, dWf, nullptr, nullptr,
                                              dBih, dBhh, dWih, oprv);
    fc_out<<<BB / 8, 256, 0, stream>>>(ho, fcW, fcb, oprv, out, s);
  }
}
